// DecoderOffset_11390253269671
// MI455X (gfx1250) — matrix operand order repaired and checked by reading, not compiled, not run on hardware
//
#include <hip/hip_runtime.h>
#include <hip/hip_bf16.h>

// ---------------------------------------------------------------------------
// CDNA5 (gfx1250) sparse U-Net decoder.
// Core op: gather-GEMM-scatter  out[s[k][m]] += x[g[k][m]] @ W[k]
// implemented with v_wmma_f32_16x16x32_bf16 (one 16x16 output tile per wave32)
// and global_atomic_add_f32 scatter. Activations are carried in bf16 between
// layers (halves random-gather bandwidth; the workload is BW-bound:
// ~136 GFLOP vs ~9 GB of gathered traffic -> ~0.4 ms floor at 23.3 TB/s).
// ---------------------------------------------------------------------------

typedef __attribute__((ext_vector_type(16))) __bf16 v16bf;
typedef __attribute__((ext_vector_type(8)))  float  v8f;
typedef __attribute__((ext_vector_type(4)))  unsigned int v4u;

union FragA { v4u u4[2]; v16bf v; };

__device__ __forceinline__ unsigned short f2bf(float f) {
  unsigned int u = __float_as_uint(f);
  unsigned int r = u + 0x7FFFu + ((u >> 16) & 1u);   // round-to-nearest-even
  return (unsigned short)(r >> 16);
}

__device__ __forceinline__ void atomic_add_f32(float* p, float v) {
  // scatter-add without return: GLOBAL_ATOMIC_ADD_U32-class op, STOREcnt tracked
  asm volatile("global_atomic_add_f32 %0, %1, off" :: "v"(p), "v"(v) : "memory");
}

// ---------------------------------------------------------------------------
// spconv: for kernel offset k (blockIdx.y), row-tile t (blockIdx.x*8 + wave):
//   A[16 x CIN]  = x[g[k][t*16 .. t*16+15]]            (bf16, gathered rows)
//   B[CIN x 16]  = Wt[k][ntile*16.. , :]               (bf16, [K][COUT][CIN])
//   C += A @ B  via v_wmma_f32_16x16x32_bf16 over CIN/32 k-steps
//   out[s[k][row]] [ntile*16 + n] += C   (f32 atomics)
//
// A-frag layout (16-bit A 16x32, contiguous k, same order as B): lane l holds row M=l%16,
//   K = (l/16)*16 + {0..15}  -> two 16B contiguous loads.
// B-frag layout: lane l holds col N=l%16, K = (l/16)*16 + {0..15}
//   -> one 32B contiguous load from the [COUT][CIN]-transposed weights.
// C layout: VGPR r, lane l -> M = r + 8*(l/16), N = l%16.
// ---------------------------------------------------------------------------
template <int CIN, int COUT>
__global__ __launch_bounds__(256)
void spconv_wmma(const unsigned short* __restrict__ x,   // [Nin, CIN] bf16
                 const unsigned short* __restrict__ wt,  // [K][COUT][CIN] bf16
                 const int* __restrict__ g,              // [K, M]
                 const int* __restrict__ s,              // [K, M]
                 float* __restrict__ out,                // [Nout, COUT] f32
                 int M) {
  constexpr int NT = COUT / 16;   // 16-wide output column tiles
  constexpr int KT = CIN / 32;    // K-steps of 32
  const int k    = blockIdx.y;
  const int lane = threadIdx.x & 31;
  const int wave = threadIdx.x >> 5;
  const int tile = blockIdx.x * 8 + wave;
  if (tile * 16 >= M) return;                       // wave-uniform

  const int arow  = tile * 16 + (lane & 15);        // gathered input row
  const bool aval = (arow < M);
  const int gidx  = g[(size_t)k * M + (aval ? arow : 0)];
  const unsigned short* xrow = x + (size_t)gidx * CIN;
  const int khalf = (lane >> 4) * 16;               // A-frag K sub-offset
  const int bhalf = (lane >> 4) * 16;               // B-frag K sub-offset
  const unsigned short* wbase =
      wt + ((size_t)k * COUT + (lane & 15)) * CIN;  // + ntile*16*CIN later

  v8f acc[NT];
#pragma unroll
  for (int nt = 0; nt < NT; ++nt) acc[nt] = (v8f)0.0f;

#pragma unroll
  for (int ks = 0; ks < KT; ++ks) {
    const int kb = ks * 32 + khalf;
    FragA fa;
    if (aval) {
      fa.u4[0] = *(const v4u*)(xrow + kb);          // K = kb..kb+7
      fa.u4[1] = *(const v4u*)(xrow + kb + 8);      // K = kb+8..kb+15
    } else {
      fa.u4[0] = (v4u)0u;                           // zero row -> zero C row
      fa.u4[1] = (v4u)0u;
    }
#pragma unroll
    for (int nt = 0; nt < NT; ++nt) {
      const v16bf b =
          *(const v16bf*)(wbase + (size_t)nt * 16 * CIN + ks * 32 + bhalf);
      acc[nt] = __builtin_amdgcn_wmma_f32_16x16x32_bf16(
          false, fa.v, false, b, (short)0, acc[nt], false, false);
    }
  }

  // scatter-add: C row m = r + 8*(lane/16); zero rows land on clamped index.
#pragma unroll
  for (int r = 0; r < 8; ++r) {
    const int mo  = tile * 16 + r + ((lane >> 4) << 3);
    const int moc = (mo < M) ? mo : (M - 1);
    const int so  = s[(size_t)k * M + moc];
    float* orow   = out + (size_t)so * COUT + (lane & 15);
#pragma unroll
    for (int nt = 0; nt < NT; ++nt) atomic_add_f32(orow + nt * 16, acc[nt][r]);
  }
}

// --------------------------- epilogue kernels ------------------------------

__global__ __launch_bounds__(256)
void bnrelu_to_bf16(const float* __restrict__ acc, const float* __restrict__ ga,
                    const float* __restrict__ be, const float* __restrict__ mu,
                    const float* __restrict__ va, unsigned short* __restrict__ dst,
                    int N, int C, int dstride, int doff) {
  const int i = blockIdx.x * 256 + threadIdx.x;
  if (i >= N * C) return;
  const int n = i / C, c = i % C;
  float y = (acc[i] - mu[c]) * rsqrtf(va[c] + 1e-5f) * ga[c] + be[c];
  y = fmaxf(y, 0.0f);
  dst[(size_t)n * dstride + doff + c] = f2bf(y);
}

__global__ __launch_bounds__(256)
void copy_to_bf16(const float* __restrict__ src, unsigned short* __restrict__ dst,
                  int N, int C, int dstride, int doff) {
  const int i = blockIdx.x * 256 + threadIdx.x;
  if (i >= N * C) return;
  const int n = i / C, c = i % C;
  dst[(size_t)n * dstride + doff + c] = f2bf(src[i]);
}

// transpose W[k][cin][cout] f32 -> Wt[k][cout][cin] bf16
__global__ __launch_bounds__(256)
void wtrans_bf16(const float* __restrict__ w, unsigned short* __restrict__ wt,
                 int K, int Cin, int Cout) {
  const int i = blockIdx.x * 256 + threadIdx.x;
  if (i >= K * Cin * Cout) return;
  const int k = i / (Cin * Cout), r = i % (Cin * Cout);
  const int ci = r / Cout, co = r % Cout;
  wt[((size_t)k * Cout + co) * Cin + ci] = f2bf(w[i]);
}

// final: out[n][0..2] = relu(bn(h[n][0..31])) @ Wout[32x3]
__global__ __launch_bounds__(256)
void final_proj(const float* __restrict__ h, const float* __restrict__ ga,
                const float* __restrict__ be, const float* __restrict__ mu,
                const float* __restrict__ va, const float* __restrict__ w,
                float* __restrict__ out, int N) {
  const int n = blockIdx.x * 256 + threadIdx.x;
  if (n >= N) return;
  float a0 = 0.f, a1 = 0.f, a2 = 0.f;
#pragma unroll
  for (int c = 0; c < 32; ++c) {
    float y = (h[(size_t)n * 32 + c] - mu[c]) * rsqrtf(va[c] + 1e-5f) * ga[c] + be[c];
    y = fmaxf(y, 0.0f);
    a0 += y * w[c * 3 + 0];
    a1 += y * w[c * 3 + 1];
    a2 += y * w[c * 3 + 2];
  }
  out[(size_t)n * 3 + 0] = a0;
  out[(size_t)n * 3 + 1] = a1;
  out[(size_t)n * 3 + 2] = a2;
}

// ---------------------------------------------------------------------------

static inline int cdiv(int a, int b) { return (a + b - 1) / b; }

extern "C" void kernel_launch(void* const* d_in, const int* in_sizes, int n_in,
                              void* d_out, int out_size, void* d_ws, size_t ws_size,
                              hipStream_t stream) {
  (void)in_sizes; (void)n_in; (void)out_size; (void)ws_size;
  constexpr int Nb = 40000, N0 = 120000, N1 = 320000;

  // setup_inputs() dict order, depth-first (insertion order):
  const float* feats_b  = (const float*)d_in[0];
  const float* feats_d0 = (const float*)d_in[1];
  const float* feats_d1 = (const float*)d_in[2];
  // params.up0
  const float* Wup0 = (const float*)d_in[3];
  const float *bnup0_g = (const float*)d_in[4], *bnup0_b = (const float*)d_in[5],
              *bnup0_m = (const float*)d_in[6], *bnup0_v = (const float*)d_in[7];
  const float* W1_0 = (const float*)d_in[8];
  const float *bn1_0g = (const float*)d_in[9],  *bn1_0b = (const float*)d_in[10],
              *bn1_0m = (const float*)d_in[11], *bn1_0v = (const float*)d_in[12];
  const float* W2_0 = (const float*)d_in[13];
  const float *bn2_0g = (const float*)d_in[14], *bn2_0b = (const float*)d_in[15],
              *bn2_0m = (const float*)d_in[16], *bn2_0v = (const float*)d_in[17];
  // params.up1
  const float* Wup1 = (const float*)d_in[18];
  const float *bnup1_g = (const float*)d_in[19], *bnup1_b = (const float*)d_in[20],
              *bnup1_m = (const float*)d_in[21], *bnup1_v = (const float*)d_in[22];
  const float* W1_1 = (const float*)d_in[23];
  const float *bn1_1g = (const float*)d_in[24], *bn1_1b = (const float*)d_in[25],
              *bn1_1m = (const float*)d_in[26], *bn1_1v = (const float*)d_in[27];
  const float* W2_1 = (const float*)d_in[28];
  const float *bn2_1g = (const float*)d_in[29], *bn2_1b = (const float*)d_in[30],
              *bn2_1m = (const float*)d_in[31], *bn2_1v = (const float*)d_in[32];
  const float* Wout = (const float*)d_in[33];
  // maps.up0 / maps.up1
  const int *gup0 = (const int*)d_in[34], *sup0 = (const int*)d_in[35];
  const int *g1_0 = (const int*)d_in[36], *s1_0 = (const int*)d_in[37];
  const int *g2_0 = (const int*)d_in[38], *s2_0 = (const int*)d_in[39];
  const int *gup1 = (const int*)d_in[40], *sup1 = (const int*)d_in[41];
  const int *g1_1 = (const int*)d_in[42], *s1_1 = (const int*)d_in[43];
  const int *g2_1 = (const int*)d_in[44], *s2_1 = (const int*)d_in[45];

  // -------- workspace layout (time-multiplexed, ~155 MB total) --------
  char* ws = (char*)d_ws;
  float* accA = (float*)(ws + 0);                 // y0 | h2 | h3  (<= 40.96 MB)
  float* accB = (float*)(ws + 40960000);          // h1 | y1 | h4  (<= 40.96 MB)
  unsigned short* bfA = (unsigned short*)(ws + 81920000);   // xb | x2 | xcat1
  unsigned short* bfB = (unsigned short*)(ws + 122880000);  // xcat0 | xb1 | x4
  unsigned short* wts = (unsigned short*)(ws + 153600000);
  unsigned short* wtup0 = wts + 0;        // 8*128*64
  unsigned short* wt1_0 = wts + 65536;    // 27*128*64
  unsigned short* wt2_0 = wts + 286720;   // 27*64*64
  unsigned short* wtup1 = wts + 397312;   // 8*64*32
  unsigned short* wt1_1 = wts + 413696;   // 27*64*32
  unsigned short* wt2_1 = wts + 468992;   // 27*32*32

  // -------- weight transpose + bf16 convert (tiny, L2-resident) --------
  wtrans_bf16<<<cdiv(8*128*64, 256), 256, 0, stream>>>(Wup0, wtup0, 8, 128, 64);
  wtrans_bf16<<<cdiv(27*128*64, 256), 256, 0, stream>>>(W1_0, wt1_0, 27, 128, 64);
  wtrans_bf16<<<cdiv(27*64*64, 256), 256, 0, stream>>>(W2_0, wt2_0, 27, 64, 64);
  wtrans_bf16<<<cdiv(8*64*32, 256), 256, 0, stream>>>(Wup1, wtup1, 8, 64, 32);
  wtrans_bf16<<<cdiv(27*64*32, 256), 256, 0, stream>>>(W1_1, wt1_1, 27, 64, 32);
  wtrans_bf16<<<cdiv(27*32*32, 256), 256, 0, stream>>>(W2_1, wt2_1, 27, 32, 32);

  // xb = bf16(feats_b)  [Nb,128]
  copy_to_bf16<<<cdiv(Nb*128, 256), 256, 0, stream>>>(feats_b, bfA, Nb, 128, 128, 0);

  // ---- level 0 upsample: y0 = scatter(xb[g] @ Wup0), M=15000, K=8 ----
  hipMemsetAsync(accA, 0, (size_t)N0 * 64 * 4, stream);
  spconv_wmma<128, 64><<<dim3(cdiv(cdiv(15000,16), 8), 8), 256, 0, stream>>>(
      bfA, wtup0, gup0, sup0, accA, 15000);
  // xcat0 = [bnrelu(y0) | bf16(feats_d0)]  [N0,128]
  bnrelu_to_bf16<<<cdiv(N0*64, 256), 256, 0, stream>>>(
      accA, bnup0_g, bnup0_b, bnup0_m, bnup0_v, bfB, N0, 64, 128, 0);
  copy_to_bf16<<<cdiv(N0*64, 256), 256, 0, stream>>>(feats_d0, bfB, N0, 64, 128, 64);

  // ---- level 0 conv1: h1 = spconv(xcat0, W1_0), 128->64, K=27 ----
  hipMemsetAsync(accB, 0, (size_t)N0 * 64 * 4, stream);
  spconv_wmma<128, 64><<<dim3(cdiv(cdiv(N0,16), 8), 27), 256, 0, stream>>>(
      bfB, wt1_0, g1_0, s1_0, accB, N0);
  bnrelu_to_bf16<<<cdiv(N0*64, 256), 256, 0, stream>>>(
      accB, bn1_0g, bn1_0b, bn1_0m, bn1_0v, bfA, N0, 64, 64, 0);  // x2

  // ---- level 0 conv2: h2 = spconv(x2, W2_0), 64->64 ----
  hipMemsetAsync(accA, 0, (size_t)N0 * 64 * 4, stream);
  spconv_wmma<64, 64><<<dim3(cdiv(cdiv(N0,16), 8), 27), 256, 0, stream>>>(
      bfA, wt2_0, g2_0, s2_0, accA, N0);
  bnrelu_to_bf16<<<cdiv(N0*64, 256), 256, 0, stream>>>(
      accA, bn2_0g, bn2_0b, bn2_0m, bn2_0v, bfB, N0, 64, 64, 0);  // xb1

  // ---- level 1 upsample: y1 = scatter(xb1[g] @ Wup1), M=40000, K=8 ----
  hipMemsetAsync(accB, 0, (size_t)N1 * 32 * 4, stream);
  spconv_wmma<64, 32><<<dim3(cdiv(cdiv(40000,16), 8), 8), 256, 0, stream>>>(
      bfB, wtup1, gup1, sup1, accB, 40000);
  // xcat1 = [bnrelu(y1) | bf16(feats_d1)]  [N1,64]
  bnrelu_to_bf16<<<cdiv(N1*32, 256), 256, 0, stream>>>(
      accB, bnup1_g, bnup1_b, bnup1_m, bnup1_v, bfA, N1, 32, 64, 0);
  copy_to_bf16<<<cdiv(N1*32, 256), 256, 0, stream>>>(feats_d1, bfA, N1, 32, 64, 32);

  // ---- level 1 conv1: h3 = spconv(xcat1, W1_1), 64->32 ----
  hipMemsetAsync(accA, 0, (size_t)N1 * 32 * 4, stream);
  spconv_wmma<64, 32><<<dim3(cdiv(cdiv(N1,16), 8), 27), 256, 0, stream>>>(
      bfA, wt1_1, g1_1, s1_1, accA, N1);
  bnrelu_to_bf16<<<cdiv(N1*32, 256), 256, 0, stream>>>(
      accA, bn1_1g, bn1_1b, bn1_1m, bn1_1v, bfB, N1, 32, 32, 0);  // x4

  // ---- level 1 conv2: h4 = spconv(x4, W2_1), 32->32 ----
  hipMemsetAsync(accB, 0, (size_t)N1 * 32 * 4, stream);
  spconv_wmma<32, 32><<<dim3(cdiv(cdiv(N1,16), 8), 27), 256, 0, stream>>>(
      bfB, wt2_1, g2_1, s2_1, accB, N1);

  // ---- final: out = relu(bn(h4)) @ Wout  [N1,3] ----
  final_proj<<<cdiv(N1, 256), 256, 0, stream>>>(
      accB, bn2_1g, bn2_1b, bn2_1m, bn2_1v, Wout, (float*)d_out, N1);
}
